// TransformerBlock_86620900425760
// MI455X (gfx1250) — compile-verified, matrix operand order checked by reading, not run on hardware
//
#include <hip/hip_runtime.h>

#define D_MODEL 64
#define N_HEADS 4
#define HEAD_DIM 16
#define D_FF 256
#define SEQ 2048
#define BATCH 4
#define ROWS (BATCH * SEQ)  // 8192

typedef _Float16 half_t;
typedef __attribute__((ext_vector_type(16))) _Float16 v16h;
typedef __attribute__((ext_vector_type(8))) _Float16 v8h;
typedef __attribute__((ext_vector_type(8))) float v8f;
typedef __attribute__((ext_vector_type(4))) unsigned int u32x4;
typedef __attribute__((ext_vector_type(8))) int i32x8;
typedef __attribute__((ext_vector_type(4))) int i32x4;

#if __has_builtin(__builtin_amdgcn_tensor_load_to_lds)
#define HAVE_TDM 1
#else
#define HAVE_TDM 0
#endif

__device__ __forceinline__ v8f wmma_f16(v16h a, v16h b, v8f c) {
  // (neg_a, A, neg_b, B, c_mod, C, reuse_a, reuse_b)
  return __builtin_amdgcn_wmma_f32_16x16x32_f16(false, a, false, b, (short)0, c,
                                                false, false);
}

#if HAVE_TDM
// 2-D tile load via the Tensor Data Mover. Descriptor per CDNA5 ISA:
// group0 = {count=1 | lds_addr | global_addr | type=2}
// group1 = {data_size, tensor_dim0/1, tile_dim0/1, tensor_dim0_stride}
__device__ __forceinline__ void tdm_load_2d(unsigned lds_off, const void* gptr,
                                            unsigned dsz_code, unsigned tile_d0,
                                            unsigned tile_d1, unsigned stride0) {
  unsigned long long ga = (unsigned long long)gptr;
  u32x4 g0;
  g0[0] = 1u;                                              // count=1
  g0[1] = lds_off;                                         // lds_addr bytes
  g0[2] = (unsigned)ga;                                    // global_addr lo
  g0[3] = ((unsigned)(ga >> 32) & 0x01FFFFFFu) | (2u << 30);  // hi | type=2
  i32x8 g1;
  g1[0] = (int)(dsz_code << 16);   // workgroup_mask=0, data_size
  g1[1] = (int)(tile_d0 << 16);    // tensor_dim0[15:0] (== tile width)
  g1[2] = (int)(tile_d1 << 16);    // tensor_dim0 hi=0 | tensor_dim1 lo
  g1[3] = (int)(tile_d0 << 16);    // tensor_dim1 hi=0 | tile_dim0
  g1[4] = (int)tile_d1;            // tile_dim1 | tile_dim2=0
  g1[5] = (int)stride0;            // tensor_dim0_stride lo32 (elements)
  g1[6] = 0;
  g1[7] = 0;
  i32x4 gz = {0, 0, 0, 0};
#if defined(__clang_major__) && __clang_major__ >= 23
  i32x8 gz8 = {0, 0, 0, 0, 0, 0, 0, 0};
  __builtin_amdgcn_tensor_load_to_lds(g0, g1, gz, gz, gz8, 0);
#else
  __builtin_amdgcn_tensor_load_to_lds(g0, g1, gz, gz, 0);
#endif
}
#endif

// ---------------------------------------------------------------------------
// Kernel 0: pre-swizzle all weights (f32 -> f16) into WMMA B-fragment order.
// Tile (kb,nt) of a [K,N] row-major matrix -> 32 lanes x 16 contiguous f16:
//   dst[(tile*32+L)*16+j] = W[(kb*32 + kmap(j, L>>4))*N + nt*16 + (L&15)]
// 96 tiles total: [0,8)Wq [8,16)Wk [16,24)Wv [24,32)Wo [32,64)W1 [64,96)W2
// ---------------------------------------------------------------------------
__global__ __launch_bounds__(128) void wswz_kernel(
    const float* __restrict__ Wq, const float* __restrict__ Wk,
    const float* __restrict__ Wv, const float* __restrict__ Wo,
    const float* __restrict__ W1, const float* __restrict__ W2,
    half_t* __restrict__ Wf) {
  const int tile = blockIdx.x * 4 + (threadIdx.x >> 5);
  const int L = threadIdx.x & 31, hs = L >> 4, i16 = L & 15;
  const float* src;
  int ld, kb, n0;
  if (tile < 24) {
    int m = tile >> 3, rem = tile & 7, h = rem >> 1;
    kb = rem & 1;
    const float* base = (m == 0) ? Wq : (m == 1) ? Wk : Wv;
    src = base + h * (D_MODEL * HEAD_DIM);
    ld = HEAD_DIM;
    n0 = 0;
  } else if (tile < 32) {
    int rem = tile - 24;
    kb = rem & 1;
    src = Wo;
    ld = D_MODEL;
    n0 = (rem >> 1) * 16;
  } else if (tile < 64) {
    int rem = tile - 32;
    kb = rem & 1;
    src = W1;
    ld = D_FF;
    n0 = (rem >> 1) * 16;
  } else {
    int rem = tile - 64;
    kb = rem & 7;
    src = W2;
    ld = D_MODEL;
    n0 = (rem >> 3) * 16;
  }
  half_t* dst = Wf + ((size_t)tile * 32 + L) * 16;
#pragma unroll
  for (int j = 0; j < 16; ++j) {
    int k = ((j < 8) ? 0 : 16) + 8 * hs + (j & 7);
    dst[j] = (half_t)src[(kb * 32 + k) * ld + n0 + i16];
  }
}

__device__ __forceinline__ v16h load_wtile(const half_t* Wf, int tile, int L) {
  return *(const v16h*)(Wf + ((size_t)tile * 32 + L) * 16);  // 32B load
}

// ---------------------------------------------------------------------------
// Kernel 1: QKV projection.  x tile via TDM -> LDS; W from fragment cache.
// Q/K written in A/B-fragment order per 16-key tile; V in B-fragment order
// per 32-key block, so the attention inner loop uses only wide loads.
//   Q/K: [(bh*128+kt)*32 + 16*(e>>3) + row%16]*8 + (e&7)
//   V:   [(bh*64 +vb)*32 + L]*16 + j  with kmap(j)=key%32, N=e
// ---------------------------------------------------------------------------
__global__ __launch_bounds__(128) void qkv_kernel(
    const float* __restrict__ x, const half_t* __restrict__ Wf,
    const float* __restrict__ bq, const float* __restrict__ bk,
    const float* __restrict__ bv, half_t* __restrict__ qswz,
    half_t* __restrict__ kswz, half_t* __restrict__ vswz) {
  __shared__ float xs[16][D_MODEL];
  const int row0 = blockIdx.x * 16;
  const int t = threadIdx.x;
#if HAVE_TDM
  if (t < 32) {
    tdm_load_2d((unsigned)(size_t)&xs[0][0], x + (size_t)row0 * D_MODEL,
                /*4B*/ 2, D_MODEL, 16, D_MODEL);
    __builtin_amdgcn_s_wait_tensorcnt(0);
  }
#else
  {
    int r = t >> 3, c0 = (t & 7) * 8;
#pragma unroll
    for (int c = 0; c < 8; ++c) xs[r][c0 + c] = x[(row0 + r) * D_MODEL + c0 + c];
  }
#endif
  __syncthreads();

  const int h = t >> 5, L = t & 31, hs = L >> 4, i16 = L & 15;
  const float* xrow = &xs[i16][0];
  v8f xv0 = *(const v8f*)(xrow + 8 * hs);
  v8f xv1 = *(const v8f*)(xrow + 16 + 8 * hs);
  v8f xv2 = *(const v8f*)(xrow + 32 + 8 * hs);
  v8f xv3 = *(const v8f*)(xrow + 48 + 8 * hs);
  v16h a0, a1;
#pragma unroll
  for (int j = 0; j < 8; ++j) {
    a0[j] = (half_t)xv0[j];
    a0[j + 8] = (half_t)xv1[j];
    a1[j] = (half_t)xv2[j];
    a1[j + 8] = (half_t)xv3[j];
  }

  const int bb = row0 >> 11;           // batch
  const int s0r = row0 & (SEQ - 1);    // seq position of tile
  const int bh = bb * N_HEADS + h;
  const int qt = s0r >> 4, vb = s0r >> 5, jb = ((s0r >> 4) & 1) * 8;
  const float* bs[3] = {bq, bk, bv};

#pragma unroll
  for (int sel = 0; sel < 3; ++sel) {
    v16h fb0 = load_wtile(Wf, sel * 8 + h * 2 + 0, L);
    v16h fb1 = load_wtile(Wf, sel * 8 + h * 2 + 1, L);
    v8f acc = {};
    acc = wmma_f16(a0, fb0, acc);
    acc = wmma_f16(a1, fb1, acc);
    float bias = bs[sel][h * HEAD_DIM + i16];
    if (sel == 2) {  // V: 8 contiguous j-slots at this lane -> one 16B store
      v8h pv;
#pragma unroll
      for (int r = 0; r < 8; ++r) pv[r] = (half_t)(acc[r] + bias);
      *(v8h*)(vswz + (((size_t)bh * 64 + vb) * 32 + L) * 16 + jb) = pv;
    } else {
      half_t* dst = (sel == 0) ? qswz : kswz;
      float scale = (sel == 0) ? 0.25f : 1.0f;  // Q pre-scaled by 1/sqrt(hd)
#pragma unroll
      for (int r = 0; r < 8; ++r) {
        int ri = r + 8 * hs;
        dst[(((size_t)bh * 128 + qt) * 32 + 16 * (i16 >> 3) + ri) * 8 +
            (i16 & 7)] = (half_t)((acc[r] + bias) * scale);
      }
    }
  }
}

// ---------------------------------------------------------------------------
// Kernel 2: causal flash attention.  wave = 16 query rows; 32-key blocks:
// 2 WMMA (scores) + online softmax + 1 WMMA (P.V); fragments are wide loads.
// ---------------------------------------------------------------------------
__global__ __launch_bounds__(128) void attn_kernel(
    const half_t* __restrict__ qswz, const half_t* __restrict__ kswz,
    const half_t* __restrict__ vswz, half_t* __restrict__ aout) {
  __shared__ half_t pbuf[4][16][32];  // per-wave P tile
  const int t = threadIdx.x, w = t >> 5, L = t & 31;
  const int hs = L >> 4, i16 = L & 15;
  const int bh = blockIdx.y;  // b*H + h
  const int bb = bh >> 2, hh = bh & (N_HEADS - 1);
  const int q0 = blockIdx.x * 64 + w * 16;

  const half_t* qh = qswz + (size_t)bh * (128 * 32 * 8);
  const half_t* kh = kswz + (size_t)bh * (128 * 32 * 8);
  const half_t* vh = vswz + (size_t)bh * (64 * 32 * 16);

  v8h qv = *(const v8h*)(qh + ((size_t)(q0 >> 4) * 32 + L) * 8);
  v16h aQ;
#pragma unroll
  for (int j = 0; j < 8; ++j) {
    aQ[j] = qv[j];                // K = 8*hs + j  (real head_dim)
    aQ[j + 8] = (half_t)0.0f;     // K >= 16 pad
  }

  float mrow[8], lrow[8];
  v8f o = {};
#pragma unroll
  for (int r = 0; r < 8; ++r) {
    mrow[r] = -1e30f;
    lrow[r] = 0.0f;
  }

  const int qend = q0 + 15;
  for (int t0 = 0; t0 <= qend; t0 += 32) {
    const int kt = t0 >> 4;
    __builtin_prefetch(kh + ((size_t)(kt + 2) * 32) * 8, 0, 3);
    v8h kv0 = *(const v8h*)(kh + ((size_t)kt * 32 + L) * 8);
    v8h kv1 = *(const v8h*)(kh + ((size_t)(kt + 1) * 32 + L) * 8);
    v16h bK0, bK1;
#pragma unroll
    for (int j = 0; j < 8; ++j) {
      bK0[j] = kv0[j];
      bK1[j] = kv1[j];
      bK0[j + 8] = (half_t)0.0f;
      bK1[j + 8] = (half_t)0.0f;
    }
    v8f z = {};
    v8f s0 = wmma_f16(aQ, bK0, z);
    v8f s1 = wmma_f16(aQ, bK1, z);

#pragma unroll
    for (int r = 0; r < 8; ++r) {
      int row = r + 8 * hs;
      int qr = q0 + row;
      float x0 = ((t0 + i16) > qr) ? -1e30f : s0[r];
      float x1 = ((t0 + 16 + i16) > qr) ? -1e30f : s1[r];
      float mx = fmaxf(x0, x1);
#pragma unroll
      for (int off = 8; off >= 1; off >>= 1)
        mx = fmaxf(mx, __shfl_xor(mx, off, 32));
      float mnew = fmaxf(mrow[r], mx);
      float alpha = __expf(mrow[r] - mnew);
      float p0 = __expf(x0 - mnew);
      float p1 = __expf(x1 - mnew);
      float rs = p0 + p1;
#pragma unroll
      for (int off = 8; off >= 1; off >>= 1) rs += __shfl_xor(rs, off, 32);
      lrow[r] = lrow[r] * alpha + rs;
      mrow[r] = mnew;
      o[r] *= alpha;
      pbuf[w][row][i16] = (half_t)p0;
      pbuf[w][row][i16 + 16] = (half_t)p1;
    }
    asm volatile("" ::: "memory");  // keep LDS P stores before reloads

    const half_t* prow = &pbuf[w][i16][0];
    v8h ap0 = *(const v8h*)(prow + 8 * hs);
    v8h ap1 = *(const v8h*)(prow + 16 + 8 * hs);
    v16h aP;
#pragma unroll
    for (int j = 0; j < 8; ++j) {
      aP[j] = ap0[j];
      aP[j + 8] = ap1[j];
    }
    v16h bV = *(const v16h*)(vh + ((size_t)(t0 >> 5) * 32 + L) * 16);
    o = wmma_f16(aP, bV, o);
  }

#pragma unroll
  for (int r = 0; r < 8; ++r) {
    int row = q0 + r + 8 * hs;
    aout[((size_t)(bb * SEQ + row)) * D_MODEL + hh * HEAD_DIM + i16] =
        (half_t)(o[r] / lrow[r]);
  }
}

// Build A fragments (K=64) from an f16 LDS row of stride `ld`.
__device__ __forceinline__ void frag_a64(const half_t* row, int hs, v16h& a0,
                                         v16h& a1) {
  v8h p0 = *(const v8h*)(row + 8 * hs);
  v8h p1 = *(const v8h*)(row + 16 + 8 * hs);
  v8h p2 = *(const v8h*)(row + 32 + 8 * hs);
  v8h p3 = *(const v8h*)(row + 48 + 8 * hs);
#pragma unroll
  for (int j = 0; j < 8; ++j) {
    a0[j] = p0[j];
    a0[j + 8] = p1[j];
    a1[j] = p2[j];
    a1[j + 8] = p3[j];
  }
}

// ---------------------------------------------------------------------------
// Kernel 3: attn @ Wo + bo, + x residual, LayerNorm1 -> h (f32 + f16)
// ---------------------------------------------------------------------------
__global__ __launch_bounds__(128) void oproj_ln1_kernel(
    const half_t* __restrict__ attn, const half_t* __restrict__ Wf,
    const float* __restrict__ bo, const float* __restrict__ x,
    const float* __restrict__ g1, const float* __restrict__ be1,
    float* __restrict__ hf32, half_t* __restrict__ hf16) {
  __shared__ half_t as[16][D_MODEL];
  __shared__ float resid[16][D_MODEL + 4];
  const int row0 = blockIdx.x * 16, t = threadIdx.x;
#if HAVE_TDM
  if (t < 32) {
    tdm_load_2d((unsigned)(size_t)&as[0][0], attn + (size_t)row0 * D_MODEL,
                /*2B*/ 1, D_MODEL, 16, D_MODEL);
    __builtin_amdgcn_s_wait_tensorcnt(0);
  }
#else
  {
    int r = t >> 3, c0 = (t & 7) * 8;
#pragma unroll
    for (int c = 0; c < 8; ++c)
      as[r][c0 + c] = attn[(row0 + r) * D_MODEL + c0 + c];
  }
#endif
  __syncthreads();

  const int w = t >> 5, L = t & 31, hs = L >> 4, i16 = L & 15;
  v16h a0, a1;
  frag_a64(&as[i16][0], hs, a0, a1);
  v16h fb0 = load_wtile(Wf, 24 + w * 2 + 0, L);
  v16h fb1 = load_wtile(Wf, 24 + w * 2 + 1, L);
  v8f acc = {};
  acc = wmma_f16(a0, fb0, acc);
  acc = wmma_f16(a1, fb1, acc);
  float bias = bo[w * 16 + i16];
#pragma unroll
  for (int r = 0; r < 8; ++r) {
    int row = r + 8 * hs;
    resid[row][w * 16 + i16] =
        acc[r] + bias + x[(row0 + row) * D_MODEL + w * 16 + i16];
  }
  __syncthreads();

  int r = t >> 3, c0 = (t & 7) * 8;
  float sum = 0.f, sq = 0.f;
#pragma unroll
  for (int c = 0; c < 8; ++c) {
    float vv = resid[r][c0 + c];
    sum += vv;
    sq += vv * vv;
  }
#pragma unroll
  for (int off = 4; off >= 1; off >>= 1) {
    sum += __shfl_xor(sum, off, 32);
    sq += __shfl_xor(sq, off, 32);
  }
  float mean = sum * (1.0f / 64.0f);
  float var = sq * (1.0f / 64.0f) - mean * mean;
  float rstd = rsqrtf(var + 1e-5f);
#pragma unroll
  for (int c = 0; c < 8; ++c) {
    int cc = c0 + c;
    float hv = (resid[r][cc] - mean) * rstd * g1[cc] + be1[cc];
    hf32[(row0 + r) * D_MODEL + cc] = hv;
    hf16[(row0 + r) * D_MODEL + cc] = (half_t)hv;
  }
}

// ---------------------------------------------------------------------------
// Kernel 4: t1 = relu(h @ W1 + b1)   [8192,64] x [64,256] -> f16
// ---------------------------------------------------------------------------
__global__ __launch_bounds__(128) void ffn1_kernel(
    const half_t* __restrict__ hf16, const half_t* __restrict__ Wf,
    const float* __restrict__ b1, half_t* __restrict__ t1) {
  __shared__ half_t hsh[16][D_MODEL];
  const int row0 = blockIdx.x * 16, t = threadIdx.x;
#if HAVE_TDM
  if (t < 32) {
    tdm_load_2d((unsigned)(size_t)&hsh[0][0], hf16 + (size_t)row0 * D_MODEL,
                1, D_MODEL, 16, D_MODEL);
    __builtin_amdgcn_s_wait_tensorcnt(0);
  }
#else
  {
    int r = t >> 3, c0 = (t & 7) * 8;
#pragma unroll
    for (int c = 0; c < 8; ++c)
      hsh[r][c0 + c] = hf16[(row0 + r) * D_MODEL + c0 + c];
  }
#endif
  __syncthreads();

  const int w = t >> 5, L = t & 31, hs = L >> 4, i16 = L & 15;
  v16h a0, a1;
  frag_a64(&hsh[i16][0], hs, a0, a1);
#pragma unroll
  for (int nt = 0; nt < 4; ++nt) {
    int ng = nt * 4 + w;  // global N tile of 16
    v16h fb0 = load_wtile(Wf, 32 + ng * 2 + 0, L);
    v16h fb1 = load_wtile(Wf, 32 + ng * 2 + 1, L);
    v8f acc = {};
    acc = wmma_f16(a0, fb0, acc);
    acc = wmma_f16(a1, fb1, acc);
    float bias = b1[ng * 16 + i16];
#pragma unroll
    for (int r = 0; r < 8; ++r) {
      int row = row0 + r + 8 * hs;
      t1[(size_t)row * D_FF + ng * 16 + i16] = (half_t)fmaxf(acc[r] + bias, 0.0f);
    }
  }
}

// ---------------------------------------------------------------------------
// Kernel 5: out = LN2(t1 @ W2 + b2 + h)   [8192,256] x [256,64] -> f32
// ---------------------------------------------------------------------------
__global__ __launch_bounds__(128) void ffn2_ln2_kernel(
    const half_t* __restrict__ t1, const half_t* __restrict__ Wf,
    const float* __restrict__ b2, const float* __restrict__ hf32,
    const float* __restrict__ g2, const float* __restrict__ be2,
    float* __restrict__ out) {
  __shared__ half_t ts[16][D_FF];
  __shared__ float resid[16][D_MODEL + 4];
  const int row0 = blockIdx.x * 16, t = threadIdx.x;
#if HAVE_TDM
  if (t < 32) {
    tdm_load_2d((unsigned)(size_t)&ts[0][0], t1 + (size_t)row0 * D_FF, 1, D_FF,
                16, D_FF);
    __builtin_amdgcn_s_wait_tensorcnt(0);
  }
#else
  {
    int r = t >> 3, c0 = (t & 7) * 32;
#pragma unroll
    for (int c = 0; c < 32; ++c)
      ts[r][c0 + c] = t1[(size_t)(row0 + r) * D_FF + c0 + c];
  }
#endif
  __syncthreads();

  const int w = t >> 5, L = t & 31, hs = L >> 4, i16 = L & 15;
  const half_t* arow = &ts[i16][0];
  v8f acc = {};
#pragma unroll
  for (int kb = 0; kb < 8; ++kb) {
    v8h p0 = *(const v8h*)(arow + kb * 32 + 8 * hs);
    v8h p1 = *(const v8h*)(arow + kb * 32 + 16 + 8 * hs);
    v16h a;
#pragma unroll
    for (int j = 0; j < 8; ++j) {
      a[j] = p0[j];
      a[j + 8] = p1[j];
    }
    v16h fb = load_wtile(Wf, 64 + w * 8 + kb, L);
    acc = wmma_f16(a, fb, acc);
  }
  float bias = b2[w * 16 + i16];
#pragma unroll
  for (int r = 0; r < 8; ++r) {
    int row = r + 8 * hs;
    resid[row][w * 16 + i16] =
        acc[r] + bias + hf32[(row0 + row) * D_MODEL + w * 16 + i16];
  }
  __syncthreads();

  int r = t >> 3, c0 = (t & 7) * 8;
  float sum = 0.f, sq = 0.f;
#pragma unroll
  for (int c = 0; c < 8; ++c) {
    float vv = resid[r][c0 + c];
    sum += vv;
    sq += vv * vv;
  }
#pragma unroll
  for (int off = 4; off >= 1; off >>= 1) {
    sum += __shfl_xor(sum, off, 32);
    sq += __shfl_xor(sq, off, 32);
  }
  float mean = sum * (1.0f / 64.0f);
  float var = sq * (1.0f / 64.0f) - mean * mean;
  float rstd = rsqrtf(var + 1e-5f);
#pragma unroll
  for (int c = 0; c < 8; ++c) {
    int cc = c0 + c;
    out[(row0 + r) * D_MODEL + cc] =
        (resid[r][cc] - mean) * rstd * g2[cc] + be2[cc];
  }
}

// ---------------------------------------------------------------------------
extern "C" void kernel_launch(void* const* d_in, const int* in_sizes, int n_in,
                              void* d_out, int out_size, void* d_ws,
                              size_t ws_size, hipStream_t stream) {
  const float* x = (const float*)d_in[0];
  const float* Wq = (const float*)d_in[1];
  const float* bq = (const float*)d_in[2];
  const float* Wk = (const float*)d_in[3];
  const float* bk = (const float*)d_in[4];
  const float* Wv = (const float*)d_in[5];
  const float* bv = (const float*)d_in[6];
  const float* Wo = (const float*)d_in[7];
  const float* bo = (const float*)d_in[8];
  const float* W1 = (const float*)d_in[9];
  const float* b1 = (const float*)d_in[10];
  const float* W2 = (const float*)d_in[11];
  const float* b2 = (const float*)d_in[12];
  const float* ln1_g = (const float*)d_in[13];
  const float* ln1_b = (const float*)d_in[14];
  const float* ln2_g = (const float*)d_in[15];
  const float* ln2_b = (const float*)d_in[16];
  float* out = (float*)d_out;

  char* ws = (char*)d_ws;
  const size_t MB = 1u << 20;
  half_t* Wf = (half_t*)(ws + 0);            // 96 tiles * 1KB = 96KB
  half_t* qswz = (half_t*)(ws + 1 * MB);     // 1MB
  half_t* kswz = (half_t*)(ws + 2 * MB);     // 1MB
  half_t* vswz = (half_t*)(ws + 3 * MB);     // 1MB
  half_t* a_ws = (half_t*)(ws + 4 * MB);     // [BS,64] f16, 1MB
  float* h32_ws = (float*)(ws + 5 * MB);     // [BS,64] f32, 2MB
  half_t* h16_ws = (half_t*)(ws + 7 * MB);   // [BS,64] f16, 1MB
  half_t* t1_ws = (half_t*)(ws + 8 * MB);    // [BS,256] f16, 4MB

  const int tiles = ROWS / 16;  // 512
  wswz_kernel<<<24, 128, 0, stream>>>(Wq, Wk, Wv, Wo, W1, W2, Wf);
  qkv_kernel<<<tiles, 128, 0, stream>>>(x, Wf, bq, bk, bv, qswz, kswz, vswz);
  attn_kernel<<<dim3(SEQ / 64, BATCH * N_HEADS), 128, 0, stream>>>(
      qswz, kswz, vswz, a_ws);
  oproj_ln1_kernel<<<tiles, 128, 0, stream>>>(a_ws, Wf, bo, x, ln1_g, ln1_b,
                                              h32_ws, h16_ws);
  ffn1_kernel<<<tiles, 128, 0, stream>>>(h16_ws, Wf, b1, t1_ws);
  ffn2_ln2_kernel<<<tiles, 128, 0, stream>>>(t1_ws, Wf, b2, h32_ws, ln2_g,
                                             ln2_b, out);
}
